// Prob_CTAttention_19756849562111
// MI455X (gfx1250) — compile-verified, matrix operand order checked by reading, not run on hardware
//
#include <hip/hip_runtime.h>
#include <hip/hip_bf16.h>

typedef __attribute__((ext_vector_type(16))) _Float16 v16h;
typedef __attribute__((ext_vector_type(8)))  float    v8f;

#define BB    4
#define LL    2048
#define HH    8
#define DD    64
#define NTOP  40
#define SAMPK 40
#define NPAD  48   // 3 row tiles of 16

static __device__ __forceinline__ unsigned hash_u32(unsigned v) {
    v = v * 747796405u + 2891336453u;
    v = ((v >> ((v >> 28) + 4u)) ^ v) * 277803737u;
    return (v >> 22) ^ v;
}

static __device__ __forceinline__ v8f wmma_f16(v16h a, v16h b, v8f c) {
    // 8 args: (neg_a, A, neg_b, B, c_mod, C, reuse_a, reuse_b)
    return __builtin_amdgcn_wmma_f32_16x16x32_f16(false, a, false, b, (short)0, c, false, false);
}

// --- gfx1250 async global->LDS copy (ASYNCcnt-tracked), per ISA 08_async_tensor ---
static __device__ __forceinline__ void async_copy_b128(unsigned lds_off, const void* gaddr) {
    asm volatile("global_load_async_to_lds_b128 %0, %1, off"
                 :: "v"(lds_off), "v"((unsigned long long)(size_t)gaddr)
                 : "memory");
}
static __device__ __forceinline__ void wait_async_le(void) {
    asm volatile("s_wait_asynccnt 0x10" ::: "memory");   // allow next chunk in flight
}
static __device__ __forceinline__ void wait_ds0(void) {
    asm volatile("s_wait_dscnt 0x0" ::: "memory");       // drain DS before async rewrites LDS
}

// Stage a 32x64 f32 tile (8KB) row-major into LDS: 512 x b128, 16 per lane.
static __device__ __forceinline__ void stage_chunk_async(unsigned lds_base,
                                                         const float* __restrict__ gsrc,
                                                         int lane) {
    #pragma unroll
    for (int i = 0; i < 16; ++i) {
        const int e = i * 32 + lane;                 // b128 index 0..511
        async_copy_b128(lds_base + (unsigned)(e * 16), gsrc + (size_t)e * 4);
    }
}

// ---------------------------------------------------------------------------
// Kernel 1: context = 0.5*((l+1)*V + cumsum(V)) written transposed (B,L,H,D)
// ---------------------------------------------------------------------------
__global__ __launch_bounds__(64) void context_kernel(const float* __restrict__ V,
                                                     float* __restrict__ out) {
    const int bh = blockIdx.x;           // b*H + h
    const int b  = bh / HH, h = bh % HH;
    const int d  = threadIdx.x;          // 0..63
    const float* Vb = V + (size_t)bh * LL * DD;
    float cs = 0.0f;
    for (int l = 0; l < LL; ++l) {
        float v = Vb[l * DD + d];
        cs += v;
        out[(((size_t)b * LL + l) * HH + h) * DD + d] = 0.5f * ((float)(l + 1) * v + cs);
    }
}

// ---------------------------------------------------------------------------
// Kernel 2: sparsity metric M[b,h,i] = max_s(q_i . k_samp) - sum_s(...)/L
// One wave32 per query row; deterministic hash sampling.
// ---------------------------------------------------------------------------
__global__ __launch_bounds__(256) void sample_kernel(const float* __restrict__ Q,
                                                     const float* __restrict__ K,
                                                     float* __restrict__ M) {
    const int wid  = (int)((blockIdx.x * blockDim.x + threadIdx.x) >> 5);
    const int lane = threadIdx.x & 31;
    if (wid >= BB * HH * LL) return;
    const int bh = wid / LL;
    const int i  = wid % LL;
    const float* Qr = Q + ((size_t)bh * LL + i) * DD;
    const float* Kb = K + (size_t)bh * LL * DD;
    const float q0 = Qr[lane], q1 = Qr[lane + 32];
    float mx = -__builtin_inff(), sum = 0.0f;
    for (int s = 0; s < SAMPK; ++s) {
        const int idx = (int)(hash_u32((unsigned)(i * SAMPK + s)) & (LL - 1));
        const float* Kr = Kb + (size_t)idx * DD;
        float p = q0 * Kr[lane] + q1 * Kr[lane + 32];
        #pragma unroll
        for (int m = 16; m >= 1; m >>= 1) p += __shfl_xor(p, m, 32);
        mx = fmaxf(mx, p);
        sum += p;
    }
    if (lane == 0) M[wid] = mx - sum * (1.0f / (float)LL);
}

// ---------------------------------------------------------------------------
// Kernel 3: top-40 of M per (b,h), descending, ties -> smaller index
// ---------------------------------------------------------------------------
__global__ __launch_bounds__(256) void topk_kernel(const float* __restrict__ M,
                                                   int* __restrict__ Mtop) {
    __shared__ float sv[LL];
    __shared__ float rv[256];
    __shared__ int   ri[256];
    const int bh = blockIdx.x, t = threadIdx.x;
    const float* Mb = M + (size_t)bh * LL;
    for (int i = t; i < LL; i += 256) sv[i] = Mb[i];
    __syncthreads();
    for (int k = 0; k < NTOP; ++k) {
        float best = -__builtin_inff(); int bi = LL;
        for (int i = t; i < LL; i += 256) {
            float v = sv[i];
            if (v > best || (v == best && i < bi)) { best = v; bi = i; }
        }
        rv[t] = best; ri[t] = bi;
        __syncthreads();
        for (int s = 128; s > 0; s >>= 1) {
            if (t < s) {
                float vo = rv[t + s]; int io = ri[t + s];
                if (vo > rv[t] || (vo == rv[t] && io < ri[t])) { rv[t] = vo; ri[t] = io; }
            }
            __syncthreads();
        }
        if (t == 0) { Mtop[bh * NTOP + k] = ri[0]; sv[ri[0]] = -__builtin_inff(); }
        __syncthreads();
    }
}

// ---------------------------------------------------------------------------
// Kernel 4: WMMA flash attention over selected rows; scatter over context.
// One wave32 per (b,h). 48 padded rows x streaming 32-column chunks.
// V chunks double-buffered in LDS via async global->LDS copies.
// ---------------------------------------------------------------------------
__global__ __launch_bounds__(32) void attn_kernel(const float* __restrict__ Q,
                                                  const float* __restrict__ K,
                                                  const float* __restrict__ V,
                                                  const int* __restrict__ Mtop,
                                                  float* __restrict__ out) {
    __shared__ _Float16 sQsel[NPAD * DD];
    __shared__ _Float16 sKsel[NPAD * DD];
    __shared__ float    sS[NPAD * 32];
    __shared__ _Float16 sP[NPAD * 32];
    __shared__ float    sV[2][32 * DD];       // async-staged V chunks
    __shared__ float    sM[NPAD], sSum[NPAD], sFac[NPAD];
    __shared__ int      sRow[NPAD];

    const int bh   = blockIdx.x;
    const int b    = bh / HH, h = bh % HH;
    const int lane = threadIdx.x & 31;
    const int lo   = lane & 15;
    const int hi   = lane >> 4;

    const float* Qb = Q + (size_t)bh * LL * DD;
    const float* Kb = K + (size_t)bh * LL * DD;
    const float* Vb = V + (size_t)bh * LL * DD;

    const unsigned vbase0 = (unsigned)(size_t)&sV[0][0];   // LDS byte offsets (addr[31:0])
    const unsigned vbase1 = (unsigned)(size_t)&sV[1][0];

    for (int u = lane; u < NPAD; u += 32) {
        int rid = (u < NTOP) ? Mtop[bh * NTOP + u] : -1;
        sRow[u] = rid;
        sM[u]   = -__builtin_inff();
        sSum[u] = 0.0f;
    }
    __syncthreads();

    // Stage selected Q/K rows as f16 (zeros for padding rows)
    for (int t = lane; t < NPAD * DD; t += 32) {
        const int u = t >> 6, d = t & 63;
        const int rid = sRow[u];
        float q = 0.0f, k = 0.0f;
        if (rid >= 0) { q = Qb[(size_t)rid * DD + d]; k = Kb[(size_t)rid * DD + d]; }
        sQsel[t] = (_Float16)q;
        sKsel[t] = (_Float16)k;
    }
    __syncthreads();

    int maxrow = 0;
    for (int u = 0; u < NTOP; ++u) maxrow = max(maxrow, sRow[u]);
    const int nchunk = (maxrow >> 5) + 1;

    v8f Oacc[3][4];
    #pragma unroll
    for (int rt = 0; rt < 3; ++rt)
        #pragma unroll
        for (int nt = 0; nt < 4; ++nt)
            #pragma unroll
            for (int r = 0; r < 8; ++r) Oacc[rt][nt][r] = 0.0f;

    const float scale = 0.5f * 0.125f;   // 0.5 * 1/sqrt(64)

    // Preload V chunk 0 into buffer 0 (async).
    stage_chunk_async(vbase0, Vb, lane);

    for (int ic = 0; ic < nchunk; ++ic) {
        const int c0  = ic << 5;
        const int buf = ic & 1;
        // Prefetch next V chunk into the other buffer (clamped redundant load on last iter).
        const int nc0 = (ic + 1 < nchunk) ? c0 + 32 : 0;
        wait_ds0();                                      // prior DS reads of that buffer done
        stage_chunk_async(buf ? vbase0 : vbase1, Vb + (size_t)nc0 * DD, lane);
        __builtin_prefetch(Kb + (size_t)(nc0 + lane) * DD, 0, 3);
        __builtin_prefetch(Qb + (size_t)(nc0 + lane) * DD, 0, 3);

        // ----- scores S[u, c0..c0+31] = Qsel.K^T + Ksel.Q^T -----
        for (int ct = 0; ct < 2; ++ct) {
            const int col = c0 + ct * 16 + lo;   // K/Q row for this B-fragment column
            v16h bK0, bK1, bQ0, bQ1;
            #pragma unroll
            for (int j = 0; j < 16; ++j) {
                const int kk = ((j >> 3) << 4) + (hi << 3) + (j & 7);
                bK0[j] = (_Float16)Kb[(size_t)col * DD + kk];
                bK1[j] = (_Float16)Kb[(size_t)col * DD + 32 + kk];
                bQ0[j] = (_Float16)Qb[(size_t)col * DD + kk];
                bQ1[j] = (_Float16)Qb[(size_t)col * DD + 32 + kk];
            }
            for (int rt = 0; rt < 3; ++rt) {
                v16h aQ0, aQ1, aK0, aK1;
                #pragma unroll
                for (int j = 0; j < 16; ++j) {
                    const int kk = ((j >> 3) << 4) + (hi << 3) + (j & 7);
                    const int r  = rt * 16 + lo;
                    aQ0[j] = sQsel[r * DD + kk];
                    aQ1[j] = sQsel[r * DD + 32 + kk];
                    aK0[j] = sKsel[r * DD + kk];
                    aK1[j] = sKsel[r * DD + 32 + kk];
                }
                v8f c;
                #pragma unroll
                for (int r = 0; r < 8; ++r) c[r] = 0.0f;
                c = wmma_f16(aQ0, bK0, c);
                c = wmma_f16(aQ1, bK1, c);
                c = wmma_f16(aK0, bQ0, c);
                c = wmma_f16(aK1, bQ1, c);
                #pragma unroll
                for (int r = 0; r < 8; ++r)
                    sS[(rt * 16 + r + 8 * hi) * 32 + ct * 16 + lo] = c[r];
            }
        }
        __syncthreads();

        // ----- online softmax update (rows distributed over lanes) -----
        for (int u = lane; u < NPAD; u += 32) {
            const int mp = sRow[u];            // allowed columns l <= mp
            const float mold = sM[u];
            float mc = -__builtin_inff();
            #pragma unroll
            for (int j = 0; j < 32; ++j) {
                float svv = sS[u * 32 + j] * scale;
                if (mp < 0 || (c0 + j) > mp) svv = -1.0e9f;
                mc = fmaxf(mc, svv);
            }
            const float mnew = fmaxf(mold, mc);
            const float f = __expf(mold - mnew);
            float ssum = sSum[u] * f;
            #pragma unroll
            for (int j = 0; j < 32; ++j) {
                float svv = sS[u * 32 + j] * scale;
                if (mp < 0 || (c0 + j) > mp) svv = -1.0e9f;
                const float e = __expf(svv - mnew);
                ssum += e;
                sP[u * 32 + j] = (_Float16)e;
            }
            sM[u] = mnew; sSum[u] = ssum; sFac[u] = f;
        }
        __syncthreads();

        // ----- rescale accumulators -----
        #pragma unroll
        for (int rt = 0; rt < 3; ++rt)
            #pragma unroll
            for (int nt = 0; nt < 4; ++nt)
                #pragma unroll
                for (int r = 0; r < 8; ++r)
                    Oacc[rt][nt][r] *= sFac[rt * 16 + r + 8 * hi];

        // ----- O += P(48x32) @ V(32x64), V gathered from async-staged LDS -----
        v16h aP[3];
        for (int rt = 0; rt < 3; ++rt) {
            #pragma unroll
            for (int j = 0; j < 16; ++j) {
                const int kk = ((j >> 3) << 4) + (hi << 3) + (j & 7);
                aP[rt][j] = sP[(rt * 16 + lo) * 32 + kk];
            }
        }
        wait_async_le();   // older 16 async b128s (current chunk) are done; next may fly
        for (int nt = 0; nt < 4; ++nt) {
            v16h bV;
            #pragma unroll
            for (int j = 0; j < 16; ++j) {
                const int kk = ((j >> 3) << 4) + (hi << 3) + (j & 7);
                bV[j] = (_Float16)sV[buf][kk * DD + nt * 16 + lo];
            }
            for (int rt = 0; rt < 3; ++rt)
                Oacc[rt][nt] = wmma_f16(aP[rt], bV, Oacc[rt][nt]);
        }
        __syncthreads();
    }

    // ----- epilogue: out[sel] = 0.5*(V_sel + attn@V) scattered over context -----
    #pragma unroll
    for (int rt = 0; rt < 3; ++rt) {
        #pragma unroll
        for (int nt = 0; nt < 4; ++nt) {
            #pragma unroll
            for (int r = 0; r < 8; ++r) {
                const int u = rt * 16 + r + 8 * hi;
                if (u < NTOP) {
                    const int rid = sRow[u];
                    if (rid >= 0) {
                        const int d = nt * 16 + lo;
                        const float o = Oacc[rt][nt][r] / sSum[u];
                        out[(((size_t)b * LL + rid) * HH + h) * DD + d] =
                            0.5f * (Vb[(size_t)rid * DD + d] + o);
                    }
                }
            }
        }
    }
}

// ---------------------------------------------------------------------------
extern "C" void kernel_launch(void* const* d_in, const int* in_sizes, int n_in,
                              void* d_out, int out_size, void* d_ws, size_t ws_size,
                              hipStream_t stream) {
    const float* Q = (const float*)d_in[0];
    const float* K = (const float*)d_in[1];
    const float* V = (const float*)d_in[2];
    float* out  = (float*)d_out;
    float* Mbuf = (float*)d_ws;                                  // B*H*L floats
    int*   Mtop = (int*)((char*)d_ws + (size_t)BB * HH * LL * sizeof(float));

    context_kernel<<<BB * HH, 64, 0, stream>>>(V, out);
    {
        const int waves  = BB * HH * LL;            // one wave32 per query row
        const int blocks = (waves * 32 + 255) / 256;
        sample_kernel<<<blocks, 256, 0, stream>>>(Q, K, Mbuf);
    }
    topk_kernel<<<BB * HH, 256, 0, stream>>>(Mbuf, Mtop);
    attn_kernel<<<BB * HH, 32, 0, stream>>>(Q, K, V, Mtop, out);
}
